// DGCNNQ_T_58643483460114
// MI455X (gfx1250) — matrix operand order repaired and checked by reading, not compiled, not run on hardware
//
#include <hip/hip_runtime.h>
#include <hip/hip_bf16.h>

// DGCNN T-Net forward for MI455X (gfx1250, wave32, WMMA).
// Heavy math: bf16 WMMA (v_wmma_f32_16x16x32_bf16) for KNN Gram matrices and
// all dense GEMM layers (2x2 register-blocked). KNN top-40 fused in 256KB LDS.
// Neighbor-index tiles staged to LDS via global_load_async_to_lds (ASYNCcnt).
// Requires ~170MB of workspace (d_ws).

#define NPTS 4096
#define BATCH 4
#define PTOT (BATCH * NPTS)   // 16384 flattened points
#define KNN 40

typedef __attribute__((ext_vector_type(16))) __bf16 v16bf;
typedef __attribute__((ext_vector_type(8)))  float  v8f;

__device__ __forceinline__ unsigned short f2bf(float f) {
    unsigned u = __builtin_bit_cast(unsigned, f);
    unsigned r = u + 0x7FFFu + ((u >> 16) & 1u);   // RNE
    return (unsigned short)(r >> 16);
}
__device__ __forceinline__ float bf2f(unsigned short h) {
    unsigned u = ((unsigned)h) << 16;
    return __builtin_bit_cast(float, u);
}

__device__ __forceinline__ v8f wmma_bf16(v16bf a, v16bf b, v8f c) {
    return __builtin_amdgcn_wmma_f32_16x16x32_bf16(
        /*neg_a=*/false, a, /*neg_b=*/false, b,
        /*c_mod=*/(short)0, c, /*reuse_a=*/false, /*reuse_b=*/false);
}

// A-matrix 16x32 bf16 fragment (ISA 7.12.2): lane M = lane&15,
// lanes<16 hold K halves {0..7,16..23}, lanes>=16 hold {8..15,24..31}.
__device__ __forceinline__ v16bf load_a_frag(const unsigned short* rowp, int lane) {
    int h8 = (lane >> 4) * 8;
    union { uint4 u[2]; v16bf v; } t;
    t.u[0] = *(const uint4*)(rowp + h8);
    t.u[1] = *(const uint4*)(rowp + 16 + h8);
    return t.v;
}
// B-matrix 32x16 bf16 fragment: lane N = lane&15, lanes<16 hold K {0..7,16..23},
// lanes>=16 hold {8..15,24..31} (same K order as A).
__device__ __forceinline__ v16bf load_b_frag(const unsigned short* rowp, int lane) {
    int h16 = (lane >> 4) * 8;
    union { uint4 u[2]; v16bf v; } t;
    t.u[0] = *(const uint4*)(rowp + h16);
    t.u[1] = *(const uint4*)(rowp + 16 + h16);
    return t.v;
}
// D-tile (16x16 f32) store: lane N = lane&15, M = (lane>>4)*8 + r, r contiguous.
__device__ __forceinline__ void store_tile(float* __restrict__ Y, int pt, int cb,
                                           int Cout, int lane, v8f acc) {
    union { v8f v; float4 f4[2]; } u; u.v = acc;
    float* yp = Y + (size_t)(pt + (lane & 15)) * Cout + cb + (lane >> 4) * 8;
    *(float4*)yp       = u.f4[0];
    *((float4*)yp + 1) = u.f4[1];
}

// ---------------------------------------------------------------------------
// Fused KNN: one workgroup = one 16-point strip of one batch.
// Phase 1: 16 waves compute the 16x4096 strip of 2*x.xT - |x_m|^2 via WMMA
//          into 256KB of dynamic LDS. A-strip fragments are hoisted into
//          registers (compile-time C) and reused across all 16 column chunks.
// Phase 2: one wave per row does iterative top-40 argmax extraction.
// ---------------------------------------------------------------------------
template <int C>
__global__ __launch_bounds__(512) void knn_topk_kernel(
    const unsigned short* __restrict__ X, int strideX,
    const float* __restrict__ xx, int* __restrict__ idxOut) {
    extern __shared__ float dist[];  // 16 * NPTS floats
    constexpr int KS = C / 32;
    int wave = threadIdx.x >> 5, lane = threadIdx.x & 31;
    int p0 = blockIdx.x * 16;        // global point base (b*N + n0)
    int b  = p0 >> 12;               // NPTS == 4096
    int rsel = lane & 15;

    // Hoist the query-strip (A) fragments: reused by all 16 chunks.
    const unsigned short* Arow = X + (size_t)(p0 + rsel) * strideX;
    v16bf afr[KS];
#pragma unroll
    for (int s = 0; s < KS; ++s) afr[s] = load_a_frag(Arow + 32 * s, lane);

    for (int chunk = 0; chunk < 16; ++chunk) {
        int m0 = chunk * 256 + wave * 16;  // column base within batch
        const unsigned short* Brow = X + (size_t)(b * NPTS + m0 + rsel) * strideX;
        v8f acc = {};
#pragma unroll
        for (int s = 0; s < KS; ++s)
            acc = wmma_bf16(afr[s], load_b_frag(Brow + 32 * s, lane), acc);
        int col = m0 + rsel;
        float xc = xx[b * NPTS + col];
        int mb = (lane >> 4) * 8;
        union { v8f v; float f[8]; } u; u.v = acc;
#pragma unroll
        for (int r = 0; r < 8; ++r)
            dist[(mb + r) * NPTS + col] = 2.f * u.f[r] - xc;
    }
    __syncthreads();

    // Phase 2: wave == strip-local row
    float* drow = dist + wave * NPTS;
    int pn = p0 + wave;
    for (int k = 0; k < KNN; ++k) {
        float best = -3.0e38f; int bi = NPTS;
        for (int j = lane; j < NPTS; j += 32) {
            float v = drow[j];
            if (v > best) { best = v; bi = j; }
        }
#pragma unroll
        for (int off = 16; off > 0; off >>= 1) {
            float ob = __shfl_xor(best, off, 32);
            int   oi = __shfl_xor(bi,   off, 32);
            if (ob > best || (ob == best && oi < bi)) { best = ob; bi = oi; }
        }
        if (lane == 0) { idxOut[(size_t)pn * KNN + k] = bi; drow[bi] = -3.0e38f; }
        __syncthreads();
    }
}

// ---------------------------------------------------------------------------
// Dense GEMM: Y[p][cout] = sum_c W[cout][c] * X[p][c]   (bf16 in, f32 out)
// 2x2 register blocking: each wave computes a 32-cout x 32-point macro-tile
// (4 accumulators); every loaded fragment feeds two WMMAs.
// grid = (Cout/32, PTOT/256), 8 waves per WG.
// ---------------------------------------------------------------------------
__global__ __launch_bounds__(256) void gemm_bf16_kernel(
    const unsigned short* __restrict__ W, const unsigned short* __restrict__ X,
    float* __restrict__ Y, int Cin, int strideX, int Cout) {
    int wave = threadIdx.x >> 5, lane = threadIdx.x & 31;
    int p0 = blockIdx.y * 256 + wave * 32;
    int cb = blockIdx.x * 32;
    int rsel = lane & 15;
    const unsigned short* A0 = W + (size_t)(cb + rsel) * Cin;
    const unsigned short* A1 = A0 + (size_t)16 * Cin;
    const unsigned short* B0 = X + (size_t)(p0 + rsel) * strideX;
    const unsigned short* B1 = B0 + (size_t)16 * strideX;
    v8f acc00 = {}, acc01 = {}, acc10 = {}, acc11 = {};
    for (int kb = 0; kb < Cin; kb += 32) {
        v16bf a0 = load_a_frag(A0 + kb, lane);
        v16bf a1 = load_a_frag(A1 + kb, lane);
        v16bf b0 = load_b_frag(B0 + kb, lane);
        v16bf b1 = load_b_frag(B1 + kb, lane);
        acc00 = wmma_bf16(a0, b0, acc00);
        acc01 = wmma_bf16(a0, b1, acc01);
        acc10 = wmma_bf16(a1, b0, acc10);
        acc11 = wmma_bf16(a1, b1, acc11);
    }
    store_tile(Y, p0,      cb,      Cout, lane, acc00);
    store_tile(Y, p0 + 16, cb,      Cout, lane, acc01);
    store_tile(Y, p0,      cb + 16, Cout, lane, acc10);
    store_tile(Y, p0 + 16, cb + 16, Cout, lane, acc11);
}

// ---------------------------------------------------------------------------
// EdgeConv gather: h[n,k,c] = y[idx[n,k]][c] + z[n][c]; track max/min over k,
// accumulate per-channel sum/sumsq (LDS reduce -> global atomics).
// The 16x40 neighbor-index tile (reused by every channel thread) is staged
// into LDS with CDNA5 async loads (ASYNCcnt).
// ---------------------------------------------------------------------------
__global__ __launch_bounds__(256) void edge_gather_kernel(
    const float* __restrict__ Y, const float* __restrict__ Z,
    const int* __restrict__ knnIdx, int C,
    float* __restrict__ hmax, float* __restrict__ hmin, float* __restrict__ stats) {
    __shared__ float ls[1024], lq[1024];
    __shared__ int sidx[16 * KNN];
    int pb = blockIdx.x * 16;
    int bbase = (pb >> 12) << 12;

    // Async-stage the index tile: direct global->LDS copy (tracked by ASYNCcnt).
    {
        const int* gidx = knnIdx + (size_t)pb * KNN;
        for (int i = threadIdx.x; i < 16 * KNN; i += 256) {
            unsigned lds = (unsigned)(size_t)&sidx[i];
            unsigned long long ga = (unsigned long long)(size_t)(gidx + i);
            asm volatile("global_load_async_to_lds_b32 %0, %1, off"
                         :: "v"(lds), "v"(ga) : "memory");
        }
    }
    for (int i = threadIdx.x; i < C; i += 256) { ls[i] = 0.f; lq[i] = 0.f; }
    asm volatile("s_wait_asynccnt 0" ::: "memory");
    __syncthreads();

    int total = 16 * C;
    for (int e = threadIdx.x; e < total; e += 256) {
        int nl = e / C, c = e - nl * C;
        int p = pb + nl;
        float z = Z[(size_t)p * C + c];
        const int* ip = &sidx[nl * KNN];
        float s = 0.f, sq = 0.f, mx = -3e38f, mn = 3e38f;
        for (int k = 0; k < KNN; ++k) {
            // Cover gather latency: one lane per 64 channels prefetches k+2.
            if ((c & 63) == 0 && k + 2 < KNN)
                __builtin_prefetch(&Y[(size_t)(bbase + ip[k + 2]) * C + c], 0, 1);
            float v = Y[(size_t)(bbase + ip[k]) * C + c] + z;
            s += v; sq += v * v; mx = fmaxf(mx, v); mn = fminf(mn, v);
        }
        hmax[(size_t)p * C + c] = mx;
        hmin[(size_t)p * C + c] = mn;
        atomicAdd(&ls[c], s); atomicAdd(&lq[c], sq);
    }
    __syncthreads();
    for (int i = threadIdx.x; i < C; i += 256) {
        atomicAdd(&stats[i], ls[i]); atomicAdd(&stats[C + i], lq[i]);
    }
}

// BN + LeakyReLU on the k-max (or k-min if gamma<0), write bf16 into cat slice.
__global__ void edge_bnact_kernel(
    const float* __restrict__ hmax, const float* __restrict__ hmin,
    const float* __restrict__ stats, const float* __restrict__ g,
    const float* __restrict__ eb, int C, unsigned short* __restrict__ dst,
    int dstStride, float invCnt, int total) {
    int t = blockIdx.x * blockDim.x + threadIdx.x;
    if (t >= total) return;
    int p = t / C, c = t - p * C;
    float mean = stats[c] * invCnt;
    float var  = stats[C + c] * invCnt - mean * mean;
    float gg = g[c];
    float sc = gg * rsqrtf(var + 1e-5f);
    float h = (gg >= 0.f ? hmax : hmin)[(size_t)p * C + c];
    float v = (h - mean) * sc + eb[c];
    v = v > 0.f ? v : 0.2f * v;
    dst[(size_t)p * dstStride + c] = f2bf(v);
}

// Per-channel sum/sumsq over all points (for w5 / t-layer BN).
__global__ __launch_bounds__(256) void colstats_kernel(
    const float* __restrict__ Y, int C, float* __restrict__ stats) {
    __shared__ float ls[1024], lq[1024];
    for (int i = threadIdx.x; i < C; i += 256) { ls[i] = 0.f; lq[i] = 0.f; }
    __syncthreads();
    size_t base = (size_t)blockIdx.x * 16 * C;
    int total = 16 * C;
    for (int e = threadIdx.x; e < total; e += 256) {
        float v = Y[base + e]; int c = e % C;
        atomicAdd(&ls[c], v); atomicAdd(&lq[c], v * v);
    }
    __syncthreads();
    for (int i = threadIdx.x; i < C; i += 256) {
        atomicAdd(&stats[i], ls[i]); atomicAdd(&stats[C + i], lq[i]);
    }
}

// BN + LeakyReLU elementwise, f32 -> bf16.
__global__ void bnact_lrelu_kernel(
    const float* __restrict__ Y, const float* __restrict__ stats,
    const float* __restrict__ g, const float* __restrict__ eb, int C,
    unsigned short* __restrict__ dst, float invCnt, int total) {
    int t = blockIdx.x * blockDim.x + threadIdx.x;
    if (t >= total) return;
    int c = t % C;
    float mean = stats[c] * invCnt;
    float var  = stats[C + c] * invCnt - mean * mean;
    float sc = g[c] * rsqrtf(var + 1e-5f);
    float v = (Y[t] - mean) * sc + eb[c];
    v = v > 0.f ? v : 0.2f * v;
    dst[t] = f2bf(v);
}

// |x|^2 per point (from bf16 activations, for the distance Gram).
__global__ void xx_kernel(const unsigned short* __restrict__ X, int C,
                          int strideX, float* __restrict__ xx, int totalP) {
    int p = blockIdx.x * blockDim.x + threadIdx.x;
    if (p >= totalP) return;
    const unsigned short* r = X + (size_t)p * strideX;
    float s = 0.f;
    for (int c = 0; c < C; ++c) { float v = bf2f(r[c]); s += v * v; }
    xx[p] = s;
}

// Per-(b,c) max & min over the 4096 points (global pool pre-pass).
__global__ void pool_minmax_kernel(const float* __restrict__ Y, int C,
                                   float* __restrict__ pmax, float* __restrict__ pmin) {
    int t = blockIdx.x * blockDim.x + threadIdx.x;
    if (t >= BATCH * C) return;
    int b = t / C, c = t - b * C;
    float mx = -3e38f, mn = 3e38f;
    for (int n = 0; n < NPTS; ++n) {
        float v = Y[(size_t)(b * NPTS + n) * C + c];
        mx = fmaxf(mx, v); mn = fminf(mn, v);
    }
    pmax[t] = mx; pmin[t] = mn;
}

__global__ void pool_bn_relu_kernel(
    const float* __restrict__ pmax, const float* __restrict__ pmin,
    const float* __restrict__ stats, const float* __restrict__ g,
    const float* __restrict__ eb, int C, float* __restrict__ out, float invCnt) {
    int t = blockIdx.x * blockDim.x + threadIdx.x;
    if (t >= BATCH * C) return;
    int c = t % C;
    float mean = stats[c] * invCnt;
    float var  = stats[C + c] * invCnt - mean * mean;
    float gg = g[c];
    float sc = gg * rsqrtf(var + 1e-5f);
    float h = (gg >= 0.f ? pmax : pmin)[t];
    out[t] = fmaxf((h - mean) * sc + eb[c], 0.f);
}

// FC + batch-BN + ReLU (batch of 4, tiny).
__global__ void fc_bn_relu_kernel(
    const float* __restrict__ in, const float* __restrict__ W,
    const float* __restrict__ g, const float* __restrict__ eb,
    int Cin, int Cout, float* __restrict__ out) {
    int c = blockIdx.x * blockDim.x + threadIdx.x;
    if (c >= Cout) return;
    float d[BATCH];
    const float* wr = W + (size_t)c * Cin;
    for (int b = 0; b < BATCH; ++b) {
        float s = 0.f;
        const float* xr = in + (size_t)b * Cin;
        for (int i = 0; i < Cin; ++i) s += xr[i] * wr[i];
        d[b] = s;
    }
    float m = (d[0] + d[1] + d[2] + d[3]) * 0.25f;
    float v = 0.f;
    for (int b = 0; b < BATCH; ++b) { float dd = d[b] - m; v += dd * dd; }
    v *= 0.25f;
    float sc = g[c] * rsqrtf(v + 1e-5f);
    for (int b = 0; b < BATCH; ++b)
        out[(size_t)b * Cout + c] = fmaxf((d[b] - m) * sc + eb[c], 0.f);
}

// Final FC (12 outs) + identity + [0,0,0,1] row -> (B,4,4).
__global__ void fc3_kernel(const float* __restrict__ in, const float* __restrict__ W,
                           const float* __restrict__ bias, float* __restrict__ out) {
    int t = threadIdx.x;
    if (t >= BATCH * 16) return;
    int b = t >> 4, i = t & 15;
    float v;
    if (i < 12) {
        float s = 0.f;
        const float* xr = in + b * 256;
        const float* wr = W + i * 256;
        for (int j = 0; j < 256; ++j) s += xr[j] * wr[j];
        const float iden[12] = {1,0,0,0, 0,1,0,0, 0,0,1,0};
        v = s + bias[i] + iden[i];
    } else {
        v = (i == 15) ? 1.f : 0.f;
    }
    out[t] = v;
}

// Weight prep: W = [Wn | Wc] (Cout x 2Cin) -> Wn (pad->CinPad), Wz = Wc - Wn.
__global__ void prep_edge_w_kernel(const float* __restrict__ w, int Cout, int Cin,
                                   int CinPad, unsigned short* __restrict__ Wn,
                                   unsigned short* __restrict__ Wz) {
    int t = blockIdx.x * blockDim.x + threadIdx.x;
    if (t >= Cout * CinPad) return;
    int o = t / CinPad, c = t - o * CinPad;
    float nv = 0.f, zv = 0.f;
    if (c < Cin) {
        nv = w[(size_t)o * 2 * Cin + c];
        zv = w[(size_t)o * 2 * Cin + Cin + c] - nv;
    }
    Wn[t] = f2bf(nv); Wz[t] = f2bf(zv);
}

__global__ void conv_bf16_kernel(const float* __restrict__ src,
                                 unsigned short* __restrict__ dst, int total) {
    int t = blockIdx.x * blockDim.x + threadIdx.x;
    if (t < total) dst[t] = f2bf(src[t]);
}

// x (B,3,N) f32 -> xT0 (B*N, 32) bf16, channels 3..31 zero-padded.
__global__ void prep_x0_kernel(const float* __restrict__ x,
                               unsigned short* __restrict__ xT0) {
    int t = blockIdx.x * blockDim.x + threadIdx.x;
    if (t >= PTOT * 32) return;
    int p = t >> 5, c = t & 31;
    int b = p >> 12, n = p & 4095;
    float v = (c < 3) ? x[((size_t)b * 3 + c) * NPTS + n] : 0.f;
    xT0[t] = f2bf(v);
}

extern "C" void kernel_launch(void* const* d_in, const int* in_sizes, int n_in,
                              void* d_out, int out_size, void* d_ws, size_t ws_size,
                              hipStream_t stream) {
    (void)in_sizes; (void)n_in; (void)out_size; (void)ws_size;
    const float* x   = (const float*)d_in[0];
    const float* w1  = (const float*)d_in[1];
    const float* g1  = (const float*)d_in[3];  const float* e1 = (const float*)d_in[4];
    const float* w2  = (const float*)d_in[5];
    const float* g2  = (const float*)d_in[7];  const float* e2 = (const float*)d_in[8];
    const float* w3  = (const float*)d_in[9];
    const float* g3  = (const float*)d_in[11]; const float* e3 = (const float*)d_in[12];
    const float* w4  = (const float*)d_in[13];
    const float* g4  = (const float*)d_in[15]; const float* e4 = (const float*)d_in[16];
    const float* w5  = (const float*)d_in[17];
    const float* g5  = (const float*)d_in[19]; const float* e5 = (const float*)d_in[20];
    const float* tw  = (const float*)d_in[21];
    const float* tg  = (const float*)d_in[23]; const float* te = (const float*)d_in[24];
    const float* f1w = (const float*)d_in[25];
    const float* f1g = (const float*)d_in[27]; const float* f1e = (const float*)d_in[28];
    const float* f2w = (const float*)d_in[29];
    const float* f2g = (const float*)d_in[31]; const float* f2e = (const float*)d_in[32];
    const float* f3w = (const float*)d_in[33]; const float* f3b = (const float*)d_in[34];

    char* ws = (char*)d_ws; size_t off = 0;
    auto alloc = [&](size_t bytes) -> char* {
        char* p = ws + off; off += (bytes + 255) & ~(size_t)255; return p;
    };
    unsigned short* catT  = (unsigned short*)alloc((size_t)PTOT * 512 * 2);
    unsigned short* featT = (unsigned short*)alloc((size_t)PTOT * 1024 * 2);
    unsigned short* xT0   = (unsigned short*)alloc((size_t)PTOT * 32 * 2);
    float* bufY  = (float*)alloc((size_t)PTOT * 1024 * 4);
    float* bufZ  = (float*)alloc((size_t)PTOT * 256 * 4);
    float* hmax  = (float*)alloc((size_t)PTOT * 256 * 4);
    float* hmin  = (float*)alloc((size_t)PTOT * 256 * 4);
    int*   knnIdx = (int*)alloc((size_t)PTOT * KNN * 4);
    float* xx    = (float*)alloc((size_t)PTOT * 4);
    float* stats = (float*)alloc(6 * 2048 * 4);       // 6 slots of 2*1024 f32
    float* pmax  = (float*)alloc(BATCH * 1024 * 4);
    float* pmin  = (float*)alloc(BATCH * 1024 * 4);
    float* hpool = (float*)alloc(BATCH * 1024 * 4);
    float* t1    = (float*)alloc(BATCH * 512 * 4);
    float* t2    = (float*)alloc(BATCH * 256 * 4);
    unsigned short* Wn1 = (unsigned short*)alloc(64 * 32 * 2);
    unsigned short* Wz1 = (unsigned short*)alloc(64 * 32 * 2);
    unsigned short* Wn2 = (unsigned short*)alloc(64 * 64 * 2);
    unsigned short* Wz2 = (unsigned short*)alloc(64 * 64 * 2);
    unsigned short* Wn3 = (unsigned short*)alloc(128 * 64 * 2);
    unsigned short* Wz3 = (unsigned short*)alloc(128 * 64 * 2);
    unsigned short* Wn4 = (unsigned short*)alloc(256 * 128 * 2);
    unsigned short* Wz4 = (unsigned short*)alloc(256 * 128 * 2);
    unsigned short* W5b = (unsigned short*)alloc(1024 * 512 * 2);
    unsigned short* Twb = (unsigned short*)alloc(1024 * 1024 * 2);

    hipMemsetAsync(stats, 0, 6 * 2048 * 4, stream);

    // Weight / input preparation (f32 -> bf16, edge factorization)
    prep_x0_kernel<<<(PTOT * 32 + 255) / 256, 256, 0, stream>>>(x, xT0);
    prep_edge_w_kernel<<<(64 * 32 + 255) / 256, 256, 0, stream>>>(w1, 64, 3, 32, Wn1, Wz1);
    prep_edge_w_kernel<<<(64 * 64 + 255) / 256, 256, 0, stream>>>(w2, 64, 64, 64, Wn2, Wz2);
    prep_edge_w_kernel<<<(128 * 64 + 255) / 256, 256, 0, stream>>>(w3, 128, 64, 64, Wn3, Wz3);
    prep_edge_w_kernel<<<(256 * 128 + 255) / 256, 256, 0, stream>>>(w4, 256, 128, 128, Wn4, Wz4);
    conv_bf16_kernel<<<(1024 * 512 + 255) / 256, 256, 0, stream>>>(w5, W5b, 1024 * 512);
    conv_bf16_kernel<<<(1024 * 1024 + 255) / 256, 256, 0, stream>>>(tw, Twb, 1024 * 1024);

    const float invEdge = 1.f / ((float)PTOT * KNN);
    const float invPts  = 1.f / (float)PTOT;

    auto launch_knn = [&](const unsigned short* X, int strideX, int Cin) {
        const size_t lds = 16 * NPTS * 4;
        if (Cin == 32)
            knn_topk_kernel<32><<<PTOT / 16, 512, lds, stream>>>(X, strideX, xx, knnIdx);
        else if (Cin == 64)
            knn_topk_kernel<64><<<PTOT / 16, 512, lds, stream>>>(X, strideX, xx, knnIdx);
        else
            knn_topk_kernel<128><<<PTOT / 16, 512, lds, stream>>>(X, strideX, xx, knnIdx);
    };

    auto edge_block = [&](const unsigned short* X, int strideX, int Cin,
                          const unsigned short* Wn, const unsigned short* Wz, int Cout,
                          const float* g, const float* e, float* statSlot,
                          unsigned short* dst) {
        xx_kernel<<<(PTOT + 255) / 256, 256, 0, stream>>>(X, Cin, strideX, xx, PTOT);
        launch_knn(X, strideX, Cin);
        dim3 gg(Cout / 32, PTOT / 256);
        gemm_bf16_kernel<<<gg, 256, 0, stream>>>(Wn, X, bufY, Cin, strideX, Cout);
        gemm_bf16_kernel<<<gg, 256, 0, stream>>>(Wz, X, bufZ, Cin, strideX, Cout);
        edge_gather_kernel<<<PTOT / 16, 256, 0, stream>>>(bufY, bufZ, knnIdx, Cout,
                                                          hmax, hmin, statSlot);
        edge_bnact_kernel<<<(PTOT * Cout + 255) / 256, 256, 0, stream>>>(
            hmax, hmin, statSlot, g, e, Cout, dst, 512, invEdge, PTOT * Cout);
    };

    // 4 EdgeConv blocks; outputs land in disjoint channel slices of catT.
    edge_block(xT0,        32,  32,  Wn1, Wz1, 64,  g1, e1, stats + 0 * 2048, catT + 0);
    edge_block(catT + 0,   512, 64,  Wn2, Wz2, 64,  g2, e2, stats + 1 * 2048, catT + 64);
    edge_block(catT + 64,  512, 64,  Wn3, Wz3, 128, g3, e3, stats + 2 * 2048, catT + 128);
    edge_block(catT + 128, 512, 128, Wn4, Wz4, 256, g4, e4, stats + 3 * 2048, catT + 256);

    // w5: 512 -> 1024, BN(0,2) + LeakyReLU
    dim3 gBig(1024 / 32, PTOT / 256);
    gemm_bf16_kernel<<<gBig, 256, 0, stream>>>(W5b, catT, bufY, 512, 512, 1024);
    colstats_kernel<<<PTOT / 16, 256, 0, stream>>>(bufY, 1024, stats + 4 * 2048);
    bnact_lrelu_kernel<<<(PTOT * 1024 + 255) / 256, 256, 0, stream>>>(
        bufY, stats + 4 * 2048, g5, e5, 1024, featT, invPts, PTOT * 1024);

    // t: 1024 -> 1024, BN(0,2) + ReLU, global max over N
    gemm_bf16_kernel<<<gBig, 256, 0, stream>>>(Twb, featT, bufY, 1024, 1024, 1024);
    colstats_kernel<<<PTOT / 16, 256, 0, stream>>>(bufY, 1024, stats + 5 * 2048);
    pool_minmax_kernel<<<(BATCH * 1024 + 255) / 256, 256, 0, stream>>>(bufY, 1024, pmax, pmin);
    pool_bn_relu_kernel<<<(BATCH * 1024 + 255) / 256, 256, 0, stream>>>(
        pmax, pmin, stats + 5 * 2048, tg, te, 1024, hpool, invPts);

    // FC head
    fc_bn_relu_kernel<<<(512 + 63) / 64, 64, 0, stream>>>(hpool, f1w, f1g, f1e, 1024, 512, t1);
    fc_bn_relu_kernel<<<(256 + 63) / 64, 64, 0, stream>>>(t1, f2w, f2g, f2e, 512, 256, t2);
    fc3_kernel<<<1, 64, 0, stream>>>(t2, f3w, f3b, (float*)d_out);
}
